// GCN_two_layers_29712583753982
// MI455X (gfx1250) — compile-verified, matrix operand order checked by reading, not run on hardware
//
#include <hip/hip_runtime.h>

// ---------- types ----------
typedef __attribute__((ext_vector_type(16))) __bf16         v16bf;
typedef __attribute__((ext_vector_type(8)))  float          v8f;
typedef __attribute__((ext_vector_type(8)))  unsigned short u16x8;
typedef __attribute__((ext_vector_type(16))) unsigned short u16x16;
typedef __attribute__((ext_vector_type(4)))  float          f32x4;

union FragU {
  v16bf  bf;
  u16x16 us;
  u16x8  h[2];
};

// HW round-to-nearest-even convert (pairs fuse to v_cvt_pk_bf16_f32).
__device__ __forceinline__ unsigned short f2bf_bits(float f) {
  __bf16 b = (__bf16)f;
  return __builtin_bit_cast(unsigned short, b);
}

// Non-temporal 16B f32 load: adj/x stream once, must not evict L2-resident B.
__device__ __forceinline__ f32x4 ldg_nt4(const float* p) {
  return __builtin_nontemporal_load((const f32x4*)p);
}

// ---------------------------------------------------------------------------
// C = A @ B (+bias) (+relu).
//   A   : M x KC row-major, f32 (AF32=1, converted on the fly w/ HW bf16 cvt)
//         or bf16 bits (AF32=0)
//   BT  : Npad x KC row-major bf16 bits  (B stored transposed)
//   MODE 0: C f32 row-major  C[m*ldc+n]
//   MODE 1: C bf16 row-major C[m*ldc+n]     (feeds next GEMM as A)
//   MODE 2: C bf16 transposed CT[n*ldc+m]   (feeds next GEMM as BT, ldc=M)
// Block: 256 threads = 8 waves arranged 4(M) x 2(N).
// Wave tile: MT 16-row tiles (stride 16) x NT 16-col tiles (stride 32).
// Block covers 64*MT rows x 32*NT cols. Invariant: Npad % (32*NT) == 0, or
// NT==1 && Npad==16 (second N-wave uniformly early-exits; EXEC stays full).
// KC is compile-time so every tile offset folds into the 24-bit IOFFSET
// (max used: 7*32*16384*2 = 7.34MB < 8MB): the K-loop body is immediate-
// offset b128 loads + two pointer bumps, minimizing WMMA->VALU WAR NOPs.
// ---------------------------------------------------------------------------
template <int MODE, bool RELU, bool AF32, int MT, int NT, int KC>
__global__ __launch_bounds__(256)
void gcn_gemm_wmma(const void* __restrict__ Aptr,
                   const unsigned short* __restrict__ BT,
                   const float* __restrict__ bias, int bias_n,
                   void* __restrict__ Cptr,
                   int M, int Npad, int ldc)
{
  const int tid  = threadIdx.x;
  const int wv   = tid >> 5;
  const int lane = tid & 31;
  const int hi   = lane >> 4;       // wave half selects the K sub-pattern
  const int r    = lane & 15;
  const int laneK = hi * 8;         // bake K phase into base pointers

  const int m0 = blockIdx.x * (64 * MT) + (wv >> 1) * (16 * MT);
  const int n0 = blockIdx.y * (32 * NT) + (wv & 1) * 16;
  if (n0 >= Npad || m0 >= M) return;            // wave-uniform branches only

  // Per-lane base pointers; all tile offsets below are compile-time consts.
  const float*          AbF = (const float*)Aptr          + (size_t)(m0 + r) * KC + laneK;
  const unsigned short* AbH = (const unsigned short*)Aptr + (size_t)(m0 + r) * KC + laneK;
  const unsigned short* Bb  = BT                          + (size_t)(n0 + r) * KC + laneK;

  v8f acc[MT][NT];
  const v8f vzero = {0.f, 0.f, 0.f, 0.f, 0.f, 0.f, 0.f, 0.f};
#pragma unroll
  for (int u = 0; u < MT; ++u)
#pragma unroll
    for (int t = 0; t < NT; ++t) acc[u][t] = vzero;

  for (int kk = 0; kk < KC; kk += 32) {
    FragU fa[MT];
#pragma unroll
    for (int u = 0; u < MT; ++u) {
      if (AF32) {
        const float* Ap = AbF + (size_t)(16 * u) * KC;   // const offset
        f32x4 a0 = ldg_nt4(Ap);
        f32x4 a1 = ldg_nt4(Ap + 4);
        f32x4 a2 = ldg_nt4(Ap + 16);
        f32x4 a3 = ldg_nt4(Ap + 20);
        if (u == 0) __builtin_prefetch(Ap + 256, 0, 0);  // global_prefetch_b8
#pragma unroll
        for (int i = 0; i < 4; ++i) {                    // -> v_cvt_pk_bf16_f32
          fa[u].bf[i]      = (__bf16)a0[i];
          fa[u].bf[4 + i]  = (__bf16)a1[i];
          fa[u].bf[8 + i]  = (__bf16)a2[i];
          fa[u].bf[12 + i] = (__bf16)a3[i];
        }
      } else {
        const unsigned short* Ap = AbH + (size_t)(16 * u) * KC;
        fa[u].h[0] = *(const u16x8*)(Ap);
        fa[u].h[1] = *(const u16x8*)(Ap + 16);
      }
    }
#pragma unroll
    for (int t = 0; t < NT; ++t) {
      const unsigned short* Bt = Bb + (size_t)(32 * t) * KC;  // const offset
      FragU fb;
      fb.h[0] = *(const u16x8*)(Bt);
      fb.h[1] = *(const u16x8*)(Bt + 16);
#pragma unroll
      for (int u = 0; u < MT; ++u) {
        acc[u][t] = __builtin_amdgcn_wmma_f32_16x16x32_bf16(
            false, fa[u].bf, false, fb.bf, (short)0, acc[u][t], false, false);
      }
    }
    AbF += 32;  AbH += 32;  Bb += 32;   // the only per-iteration address VALU
  }

  // Epilogue: bias + relu + convert/store.
  // C tile layout: VGPR i holds M = m0+16u+hi*8+i, N = n0+32t+r.
#pragma unroll
  for (int t = 0; t < NT; ++t) {
    const int n = n0 + 32 * t + r;
    const float bv = (bias != nullptr && n < bias_n) ? bias[n] : 0.f;
#pragma unroll
    for (int u = 0; u < MT; ++u) {
      const int mbase = m0 + 16 * u + hi * 8;
      if (MODE == 2) {
        unsigned short* CT = (unsigned short*)Cptr;
        u16x8 pack;
#pragma unroll
        for (int i = 0; i < 8; ++i) {
          float v = acc[u][t][i] + bv;
          if (RELU) v = fmaxf(v, 0.f);
          pack[i] = f2bf_bits(v);
        }
        *(u16x8*)(CT + (size_t)n * ldc + mbase) = pack;      // 16B store
      } else if (MODE == 1) {
        unsigned short* C = (unsigned short*)Cptr;
#pragma unroll
        for (int i = 0; i < 8; ++i) {
          float v = acc[u][t][i] + bv;
          if (RELU) v = fmaxf(v, 0.f);
          C[(size_t)(mbase + i) * ldc + n] = f2bf_bits(v);
        }
      } else {
        float* C = (float*)Cptr;
#pragma unroll
        for (int i = 0; i < 8; ++i) {
          float v = acc[u][t][i] + bv;
          if (RELU) v = fmaxf(v, 0.f);
          C[(size_t)(mbase + i) * ldc + n] = v;
        }
      }
    }
  }
}

// W (K x N, f32) -> WT (Npad x K, bf16 bits), zero-padded rows n >= N.
__global__ void gcn_transpose_pad_bf16(const float* __restrict__ W,
                                       unsigned short* __restrict__ WT,
                                       int K, int N, int Npad)
{
  int idx = blockIdx.x * blockDim.x + threadIdx.x;
  if (idx >= Npad * K) return;
  int nn = idx / K;
  int kk = idx - nn * K;
  float v = (nn < N) ? W[(size_t)kk * N + nn] : 0.f;
  WT[idx] = f2bf_bits(v);
}

// Row-wise log_softmax over first 8 of 16 padded columns.
__global__ void gcn_logsoftmax8(const float* __restrict__ H,
                                float* __restrict__ out, int M)
{
  int row = blockIdx.x * blockDim.x + threadIdx.x;
  if (row >= M) return;
  const float* h = H + (size_t)row * 16;
  float m = h[0];
#pragma unroll
  for (int i = 1; i < 8; ++i) m = fmaxf(m, h[i]);
  float s = 0.f;
#pragma unroll
  for (int i = 0; i < 8; ++i) s += expf(h[i] - m);
  float ls = m + logf(s);
#pragma unroll
  for (int i = 0; i < 8; ++i) out[(size_t)row * 8 + i] = h[i] - ls;
}

extern "C" void kernel_launch(void* const* d_in, const int* in_sizes, int n_in,
                              void* d_out, int out_size, void* d_ws, size_t ws_size,
                              hipStream_t stream)
{
  (void)in_sizes; (void)n_in; (void)out_size; (void)ws_size;
  const float* x   = (const float*)d_in[0];
  const float* adj = (const float*)d_in[1];
  const float* W1  = (const float*)d_in[2];
  const float* b1  = (const float*)d_in[3];
  const float* W2  = (const float*)d_in[4];
  const float* b2  = (const float*)d_in[5];
  const float* W3  = (const float*)d_in[6];
  const float* b3  = (const float*)d_in[7];

  const int N = 16384, F = 512, H1 = 256, H2 = 128, C = 8, Cp = 16;

  char* ws = (char*)d_ws;
  size_t off = 0;
  auto alloc = [&](size_t bytes) -> void* {
    void* p = ws + off;
    off += (bytes + 255) & ~(size_t)255;
    return p;
  };
  unsigned short* W1T = (unsigned short*)alloc((size_t)H1 * F  * 2); // 256x512
  unsigned short* W2T = (unsigned short*)alloc((size_t)H2 * H1 * 2); // 128x256
  unsigned short* W3T = (unsigned short*)alloc((size_t)Cp * H2 * 2); // 16x128, rows 8..15 = 0
  unsigned short* s1T = (unsigned short*)alloc((size_t)H1 * N  * 2); // (x@W1)^T
  unsigned short* h1  = (unsigned short*)alloc((size_t)N  * H1 * 2); // relu layer 1
  unsigned short* s2T = (unsigned short*)alloc((size_t)H2 * N  * 2); // (h1@W2)^T
  unsigned short* h2  = (unsigned short*)alloc((size_t)N  * H2 * 2); // relu layer 2
  unsigned short* s3T = (unsigned short*)alloc((size_t)Cp * N  * 2); // (h2@W3)^T padded
  float*          h3  = (float*)         alloc((size_t)N  * Cp * 4); // pre-softmax logits

  dim3 blk(256);

  // Weight transposes (tiny, L2-resident).
  gcn_transpose_pad_bf16<<<dim3((H1 * F  + 255) / 256), blk, 0, stream>>>(W1, W1T, F,  H1, H1);
  gcn_transpose_pad_bf16<<<dim3((H2 * H1 + 255) / 256), blk, 0, stream>>>(W2, W2T, H1, H2, H2);
  gcn_transpose_pad_bf16<<<dim3((Cp * H2 + 255) / 256), blk, 0, stream>>>(W3, W3T, H2, C,  Cp);

  // Layer 1: s1 = x @ W1 (store transposed), h1 = relu(adj @ s1 + b1)
  gcn_gemm_wmma<2, false, true, 2, 8, 512><<<dim3(N / 128, 1), blk, 0, stream>>>(
      x, W1T, nullptr, 0, s1T, N, H1, N);
  gcn_gemm_wmma<1, true, true, 2, 8, 16384><<<dim3(N / 128, 1), blk, 0, stream>>>(
      adj, s1T, b1, H1, h1, N, H1, H1);

  // Layer 2: s2 = h1 @ W2 (transposed), h2 = relu(adj @ s2 + b2)
  gcn_gemm_wmma<2, false, false, 2, 4, 256><<<dim3(N / 128, 1), blk, 0, stream>>>(
      h1, W2T, nullptr, 0, s2T, N, H2, N);
  gcn_gemm_wmma<1, true, true, 2, 4, 16384><<<dim3(N / 128, 1), blk, 0, stream>>>(
      adj, s2T, b2, H2, h2, N, H2, H2);

  // Layer 3 (padded to 16 cols): s3 = h2 @ W3 (transposed), h3 = adj @ s3 + b3
  gcn_gemm_wmma<2, false, false, 2, 1, 128><<<dim3(N / 128, 1), blk, 0, stream>>>(
      h2, W3T, nullptr, 0, s3T, N, Cp, N);
  gcn_gemm_wmma<0, false, true, 2, 1, 16384><<<dim3(N / 128, 1), blk, 0, stream>>>(
      adj, s3T, b3, C, h3, N, Cp, Cp);

  // log_softmax over the 8 real classes.
  gcn_logsoftmax8<<<dim3((N + 255) / 256), blk, 0, stream>>>(h3, (float*)d_out, N);
}
